// WanI2VCrossAttention_15599321219099
// MI455X (gfx1250) — matrix operand order repaired and checked by reading, not compiled, not run on hardware
//
#include <hip/hip_runtime.h>
#include <hip/hip_bf16.h>

typedef __bf16 bf16_t;
typedef __bf16 v16bf __attribute__((ext_vector_type(16)));
typedef float  v8f   __attribute__((ext_vector_type(8)));

#define DIM     1536
#define HEADS   12
#define HD      128
#define SEQP    776      // per-head KV row/col stride (img 0..256, pad, txt 264..775)
#define TXTBASE 264
#define IMGLEN  257
#define TXTLEN  512
#define LQ      16384
#define CTXLEN  769

__device__ __forceinline__ v8f zero_v8() {
  v8f z; for (int r = 0; r < 8; r++) z[r] = 0.0f; return z;
}

// 16 consecutive bf16 (32B) -> B fragment
__device__ __forceinline__ v16bf ld16(const bf16_t* p) {
  union { v16bf v; uint4 u[2]; } f;
  f.u[0] = *(const uint4*)(p);
  f.u[1] = *(const uint4*)(p + 8);
  return f.v;
}
// A fragment: elements [half*16 .. +15] of a 32-wide k window (contiguous)
__device__ __forceinline__ v16bf ldA(const bf16_t* row, int half) {
  union { v16bf v; uint4 u[2]; } f;
  f.u[0] = *(const uint4*)(row + half * 16);
  f.u[1] = *(const uint4*)(row + half * 16 + 8);
  return f.v;
}

#define WMMA_BF16(a, b, c) \
  __builtin_amdgcn_wmma_f32_16x16x32_bf16(false, (a), false, (b), (short)0, (c), false, false)

// ---------------------------------------------------------------------------
// C[M x 1536](f32) = A[M x 1536](bf16) @ W^T(bf16, stored [N][K]) + bias
// 128x128 tile per WG, 8 waves: wave -> (2 m-groups of 64) x (4 n-groups of 32)
// A tile staged to LDS with CDNA5 async global->LDS copies (ASYNCcnt path).
// ---------------------------------------------------------------------------
__global__ __launch_bounds__(256) void gemm_bf16(
    const bf16_t* __restrict__ A, int M,
    const bf16_t* __restrict__ Wt,
    const float* __restrict__ bias,
    float* __restrict__ C)
{
  __shared__ bf16_t As[128][40];   // 32-wide k tile, padded (80B row, 16B aligned)

  const int tid  = threadIdx.x;
  const int wave = tid >> 5, lane = tid & 31;
  const int half = lane >> 4, ln = lane & 15;
  const int wm = (wave & 1) * 64, wn = (wave >> 1) * 32;
  const int bm = blockIdx.x * 128, bn = blockIdx.y * 128;

  v8f acc[4][2];
  for (int i = 0; i < 4; i++) for (int j = 0; j < 2; j++) acc[i][j] = zero_v8();

  // staging assignment: thread -> (row, 16-col half) of the 128x32 A tile
  const int srow = tid >> 1;
  const int scol = (tid & 1) * 16;
  int gr = bm + srow; if (gr >= M) gr = M - 1;
  const unsigned ldsaddr = (unsigned)(unsigned long long)&As[srow][scol];
  const unsigned gbase   = (unsigned)(((size_t)gr * DIM + scol) * 2u);

  for (int k0 = 0; k0 < DIM; k0 += 32) {
    __syncthreads();
    // async copy 32B/lane: global (SADDR + voffset) -> LDS, tracked by ASYNCcnt
    unsigned voff = gbase + (unsigned)(k0 * 2);
    asm volatile(
        "global_load_async_to_lds_b128 %0, %1, %2\n\t"
        "global_load_async_to_lds_b128 %0, %1, %2 offset:16"
        :: "v"(ldsaddr), "v"(voff), "s"(A) : "memory");
    asm volatile("s_wait_asynccnt 0x0" ::: "memory");
    __syncthreads();

    for (int nn = 0; nn < 2; nn++) {
      int n = bn + wn + nn * 16 + ln;
      v16bf b = ld16(Wt + (size_t)n * DIM + k0 + half * 16);
      for (int mm = 0; mm < 4; mm++) {
        v16bf a = ldA(&As[wm + mm * 16 + ln][0], half);
        acc[mm][nn] = WMMA_BF16(a, b, acc[mm][nn]);
      }
    }
  }

  const int n0 = bn + wn + ln;
  const float bv0 = bias[n0];
  const float bv1 = bias[n0 + 16];
  if (bm + 127 < M) {           // fast path: full tile, no guards
    for (int mm = 0; mm < 4; mm++) {
      size_t mrow = (size_t)(bm + wm + mm * 16 + half * 8) * DIM;
      for (int r = 0; r < 8; r++) {
        C[mrow + (size_t)r * DIM + n0]      = acc[mm][0][r] + bv0;
        C[mrow + (size_t)r * DIM + n0 + 16] = acc[mm][1][r] + bv1;
      }
    }
  } else {
    for (int mm = 0; mm < 4; mm++)
      for (int r = 0; r < 8; r++) {
        int m = bm + wm + mm * 16 + half * 8 + r;
        if (m < M) {
          C[(size_t)m * DIM + n0]      = acc[mm][0][r] + bv0;
          C[(size_t)m * DIM + n0 + 16] = acc[mm][1][r] + bv1;
        }
      }
  }
}

// ---------------------------------------------------------------------------
// prep / epilogue kernels
// ---------------------------------------------------------------------------
__global__ __launch_bounds__(256) void conv_bf16x4(const float* __restrict__ X,
                                                   bf16_t* __restrict__ Y, int n4)
{
  int i = blockIdx.x * 256 + threadIdx.x;
  if (i >= n4) return;
  float4 f = ((const float4*)X)[i];
  union { bf16_t h[4]; uint2 u; } p;
  p.h[0] = (bf16_t)f.x; p.h[1] = (bf16_t)f.y; p.h[2] = (bf16_t)f.z; p.h[3] = (bf16_t)f.w;
  *(uint2*)&Y[(size_t)i * 4] = p.u;
}

__global__ __launch_bounds__(256) void wtrans(const float* __restrict__ W,
                                              bf16_t* __restrict__ Wt)
{
  size_t id = (size_t)blockIdx.x * 256 + threadIdx.x;
  if (id >= (size_t)DIM * DIM) return;
  size_t n = id / DIM, k = id % DIM;
  Wt[id] = (bf16_t)W[k * DIM + n];
}

__global__ __launch_bounds__(256) void zero32(unsigned* p, int n)
{
  int i = blockIdx.x * 256 + threadIdx.x;
  if (i < n) p[i] = 0u;
}

__global__ __launch_bounds__(256) void rmsnorm_rows(
    const float* __restrict__ X, const float* __restrict__ g,
    bf16_t* __restrict__ Y)
{
  int row = blockIdx.x;
  const float* xr = X + (size_t)row * DIM;
  __shared__ float red[256];
  float s = 0.f;
  for (int i = threadIdx.x; i < DIM; i += 256) { float v = xr[i]; s += v * v; }
  red[threadIdx.x] = s; __syncthreads();
  for (int o = 128; o > 0; o >>= 1) {
    if (threadIdx.x < o) red[threadIdx.x] += red[threadIdx.x + o];
    __syncthreads();
  }
  float inv = rsqrtf(red[0] / (float)DIM + 1e-6f);
  for (int i = threadIdx.x; i < DIM; i += 256)
    Y[(size_t)row * DIM + i] = (bf16_t)(xr[i] * inv * g[i]);
}

__global__ __launch_bounds__(256) void rmsnorm_scatter_k(
    const float* __restrict__ X, const float* __restrict__ g,
    bf16_t* __restrict__ Kws, int baserow)
{
  int s = blockIdx.x;
  const float* xr = X + (size_t)s * DIM;
  __shared__ float red[256];
  float a = 0.f;
  for (int i = threadIdx.x; i < DIM; i += 256) { float v = xr[i]; a += v * v; }
  red[threadIdx.x] = a; __syncthreads();
  for (int o = 128; o > 0; o >>= 1) {
    if (threadIdx.x < o) red[threadIdx.x] += red[threadIdx.x + o];
    __syncthreads();
  }
  float inv = rsqrtf(red[0] / (float)DIM + 1e-6f);
  for (int i = threadIdx.x; i < DIM; i += 256) {
    int hh = i >> 7, d = i & 127;
    Kws[((size_t)hh * SEQP + baserow + s) * HD + d] = (bf16_t)(xr[i] * inv * g[i]);
  }
}

__global__ __launch_bounds__(256) void scatter_vt(
    const float* __restrict__ X, bf16_t* __restrict__ Vt, int basecol)
{
  int s = blockIdx.x;
  for (int i = threadIdx.x; i < DIM; i += 256) {
    int hh = i >> 7, d = i & 127;
    Vt[((size_t)hh * HD + d) * SEQP + basecol + s] = (bf16_t)X[(size_t)s * DIM + i];
  }
}

// ---------------------------------------------------------------------------
// Flash-style cross attention: block = 128 queries x 1 head, wave = 16 queries.
// Streams KV in 32-wide chunks; separate softmax per segment (img / txt),
// results summed. Output written bf16 (consumed by bf16 out-projection GEMM).
// ---------------------------------------------------------------------------
__global__ __launch_bounds__(256) void attn_flash(
    const bf16_t* __restrict__ Qb,   // [LQ][DIM] bf16 (rmsnormed)
    const bf16_t* __restrict__ Kws,  // [HEADS*SEQP][HD]
    const bf16_t* __restrict__ Vt,   // [HEADS*HD][SEQP]
    bf16_t* __restrict__ O)          // [LQ][DIM]
{
  __shared__ bf16_t Qs[128][136];
  __shared__ bf16_t Pst[8][16][40];

  const int tid  = threadIdx.x;
  const int wave = tid >> 5, lane = tid & 31;
  const int half = lane >> 4, ln = lane & 15;
  const int qb = blockIdx.x, h = blockIdx.y;
  const float scale = 0.08838834764831845f;  // 1/sqrt(128)

  { // load Q block (128 x 128 bf16) into LDS
    int row = tid >> 1;
    int c = (tid & 1) * 64;
    const uint4* s4 = (const uint4*)(Qb + (size_t)(qb * 128 + row) * DIM + h * HD + c);
    uint4* dst = (uint4*)&Qs[row][c];
    for (int i = 0; i < 8; i++) dst[i] = s4[i];
  }
  __syncthreads();

  const bf16_t* qrow = &Qs[wave * 16 + ln][0];

  v8f ofin[8];
  for (int nt = 0; nt < 8; nt++) ofin[nt] = zero_v8();

  for (int seg = 0; seg < 2; seg++) {
    const int base = seg ? TXTBASE : 0;
    const int len  = seg ? TXTLEN : IMGLEN;
    const int nch  = seg ? (TXTLEN / 32) : ((IMGLEN + 31) / 32);
    const bf16_t* kbase = Kws + (size_t)(h * SEQP + base) * HD;
    const bf16_t* vbase = Vt + (size_t)(h * HD) * SEQP + base;

    v8f oacc[8];
    for (int nt = 0; nt < 8; nt++) oacc[nt] = zero_v8();
    float rmax[8], rsum[8];
    for (int r = 0; r < 8; r++) { rmax[r] = -3.0e38f; rsum[r] = 0.f; }

    for (int ch = 0; ch < nch; ch++) {
      int cb = ch * 32;
      // S tile (16 x 32) = Q x K^T
      v8f s0 = zero_v8(), s1 = zero_v8();
      for (int k0 = 0; k0 < HD; k0 += 32) {
        v16bf a  = ldA(qrow + k0, half);
        v16bf b0 = ld16(kbase + (size_t)(cb + ln) * HD + k0 + half * 16);
        v16bf b1 = ld16(kbase + (size_t)(cb + 16 + ln) * HD + k0 + half * 16);
        s0 = WMMA_BF16(a, b0, s0);
        s1 = WMMA_BF16(a, b1, s1);
      }
      // online softmax (row = half*8 + r, spread over the 16 'ln' lanes)
      bool v0 = (cb + ln) < len, v1 = (cb + 16 + ln) < len;
      float corr[8];
      for (int r = 0; r < 8; r++) {
        float x0 = v0 ? s0[r] * scale : -3.0e38f;
        float x1 = v1 ? s1[r] * scale : -3.0e38f;
        float cm = fmaxf(x0, x1);
        for (int m = 1; m < 16; m <<= 1) cm = fmaxf(cm, __shfl_xor(cm, m, 32));
        float nm = fmaxf(rmax[r], cm);
        corr[r] = __expf(rmax[r] - nm);
        rmax[r] = nm;
        float e0 = v0 ? __expf(x0 - nm) : 0.f;
        float e1 = v1 ? __expf(x1 - nm) : 0.f;
        float cs = e0 + e1;
        for (int m = 1; m < 16; m <<= 1) cs += __shfl_xor(cs, m, 32);
        rsum[r] = rsum[r] * corr[r] + cs;
        Pst[wave][half * 8 + r][ln]      = (bf16_t)e0;
        Pst[wave][half * 8 + r][16 + ln] = (bf16_t)e1;
      }
      for (int nt = 0; nt < 8; nt++)
        for (int r = 0; r < 8; r++)
          oacc[nt][r] *= corr[r];
      // O += P x V_chunk (K = 32, one WMMA step per n-tile)
      v16bf ap = ldA(&Pst[wave][ln][0], half);
      for (int nt = 0; nt < 8; nt++) {
        v16bf bv = ld16(vbase + (size_t)(nt * 16 + ln) * SEQP + cb + half * 16);
        oacc[nt] = WMMA_BF16(ap, bv, oacc[nt]);
      }
    }
    for (int r = 0; r < 8; r++) {
      float inv = 1.0f / rsum[r];
      for (int nt = 0; nt < 8; nt++) ofin[nt][r] += oacc[nt][r] * inv;
    }
  }

  for (int nt = 0; nt < 8; nt++)
    for (int r = 0; r < 8; r++) {
      int m = qb * 128 + wave * 16 + half * 8 + r;
      O[(size_t)m * DIM + h * HD + nt * 16 + ln] = (bf16_t)ofin[nt][r];
    }
}

// ---------------------------------------------------------------------------
extern "C" void kernel_launch(void* const* d_in, const int* in_sizes, int n_in,
                              void* d_out, int out_size, void* d_ws, size_t ws_size,
                              hipStream_t stream)
{
  const float* x    = (const float*)d_in[0];
  const float* ctx  = (const float*)d_in[1];
  // d_in[2] = context_lens (unused by reference)
  const float* Wq  = (const float*)d_in[3];  const float* bq  = (const float*)d_in[4];
  const float* Wk  = (const float*)d_in[5];  const float* bk  = (const float*)d_in[6];
  const float* Wv  = (const float*)d_in[7];  const float* bv  = (const float*)d_in[8];
  const float* Wak = (const float*)d_in[9];  const float* bak = (const float*)d_in[10];
  const float* Wav = (const float*)d_in[11]; const float* bav = (const float*)d_in[12];
  const float* Wo  = (const float*)d_in[13]; const float* bo  = (const float*)d_in[14];
  const float* g_q = (const float*)d_in[15];
  const float* g_k = (const float*)d_in[16];
  const float* g_ak= (const float*)d_in[17];

  char* ws = (char*)d_ws;
  float*  tmp  = (float*)(ws);                      // 16384x1536 f32 (Qproj out); later reused as bf16 attn out
  bf16_t* Ob   = (bf16_t*)(ws);                     // bf16 attention output (overlays tmp, used after tmp consumed)
  bf16_t* Qb   = (bf16_t*)(ws + 100663296);         // 50.3MB
  bf16_t* xb   = (bf16_t*)(ws + 150994944);         // x in bf16, 50.3MB
  bf16_t* ctxb = (bf16_t*)(ws + 201326592);         // context in bf16, 2.36MB
  bf16_t* Wt0  = (bf16_t*)(ws + 203688960);         // 6 x 4.72MB transposed bf16 weights
  bf16_t* Wt1  = (bf16_t*)(ws + 203688960 + 1 * 4718592);
  bf16_t* Wt2  = (bf16_t*)(ws + 203688960 + 2 * 4718592);
  bf16_t* Wt3  = (bf16_t*)(ws + 203688960 + 3 * 4718592);
  bf16_t* Wt4  = (bf16_t*)(ws + 203688960 + 4 * 4718592);
  bf16_t* Wt5  = (bf16_t*)(ws + 203688960 + 5 * 4718592);
  bf16_t* Kws  = (bf16_t*)(ws + 232000512);         // 12*776*128 bf16
  bf16_t* Vt   = (bf16_t*)(ws + 234384384);         // 12*128*776 bf16
  float*  ktmp = (float*)(ws + 236768256);          // 512x1536 f32

  // zero K/V pad regions (Kws + Vt contiguous: 4,767,744 B = 1,191,936 dwords)
  zero32<<<(1191936 + 255) / 256, 256, 0, stream>>>((unsigned*)Kws, 1191936);

  // activations -> bf16
  {
    int n4 = LQ * DIM / 4;
    conv_bf16x4<<<(n4 + 255) / 256, 256, 0, stream>>>(x, xb, n4);
    int c4 = CTXLEN * DIM / 4;
    conv_bf16x4<<<(c4 + 255) / 256, 256, 0, stream>>>(ctx, ctxb, c4);
  }

  // weight transpose + bf16 convert
  int wg = (DIM * DIM + 255) / 256;
  wtrans<<<wg, 256, 0, stream>>>(Wq,  Wt0);
  wtrans<<<wg, 256, 0, stream>>>(Wk,  Wt1);
  wtrans<<<wg, 256, 0, stream>>>(Wv,  Wt2);
  wtrans<<<wg, 256, 0, stream>>>(Wak, Wt3);
  wtrans<<<wg, 256, 0, stream>>>(Wav, Wt4);
  wtrans<<<wg, 256, 0, stream>>>(Wo,  Wt5);

  const bf16_t* ctxb_img = ctxb;
  const bf16_t* ctxb_txt = ctxb + (size_t)IMGLEN * DIM;

  // Q projection + rmsnorm -> bf16
  dim3 gbig(LQ / 128, DIM / 128);
  gemm_bf16<<<gbig, 256, 0, stream>>>(xb, LQ, Wt0, bq, tmp);
  rmsnorm_rows<<<LQ, 256, 0, stream>>>(tmp, g_q, Qb);

  // txt K / V
  dim3 gtxt((TXTLEN + 127) / 128, DIM / 128);
  gemm_bf16<<<gtxt, 256, 0, stream>>>(ctxb_txt, TXTLEN, Wt1, bk, ktmp);
  rmsnorm_scatter_k<<<TXTLEN, 256, 0, stream>>>(ktmp, g_k, Kws, TXTBASE);
  gemm_bf16<<<gtxt, 256, 0, stream>>>(ctxb_txt, TXTLEN, Wt2, bv, ktmp);
  scatter_vt<<<TXTLEN, 256, 0, stream>>>(ktmp, Vt, TXTBASE);

  // img K / V
  dim3 gimg((IMGLEN + 127) / 128, DIM / 128);
  gemm_bf16<<<gimg, 256, 0, stream>>>(ctxb_img, IMGLEN, Wt3, bak, ktmp);
  rmsnorm_scatter_k<<<IMGLEN, 256, 0, stream>>>(ktmp, g_ak, Kws, 0);
  gemm_bf16<<<gimg, 256, 0, stream>>>(ctxb_img, IMGLEN, Wt4, bav, ktmp);
  scatter_vt<<<IMGLEN, 256, 0, stream>>>(ktmp, Vt, 0);

  // attention (img + txt, separate softmax, summed) -> Ob (bf16, overlays tmp)
  dim3 ga(LQ / 128, HEADS);
  attn_flash<<<ga, 256, 0, stream>>>(Qb, Kws, Vt, Ob);

  // output projection -> d_out (f32)
  gemm_bf16<<<gbig, 256, 0, stream>>>(Ob, LQ, Wt5, bo, (float*)d_out);
}
